// SiameseGNN_SAGE_31954556682876
// MI455X (gfx1250) — matrix operand order repaired and checked by reading, not compiled, not run on hardware
//
#include <hip/hip_runtime.h>
#include <hip/hip_bf16.h>
#include <math.h>

typedef __attribute__((ext_vector_type(16))) __bf16 v16bf;
typedef __attribute__((ext_vector_type(8)))  float  v8f;

#define F_IN 128
#define H1DIM 128
#define H2DIM 64
#define NGRAPH 10
#define NPG 5000
#define TOPK 50

// ---------------------------------------------------------------- utilities
__global__ void zero_kernel(float* __restrict__ p, long long n) {
    long long i = (long long)blockIdx.x * blockDim.x + threadIdx.x;
    long long stride = (long long)gridDim.x * blockDim.x;
    for (; i < n; i += stride) p[i] = 0.0f;
}

// ------------------------------------------------- mean-aggregation scatter
// one thread per (edge, float4 chunk); chunk==0 thread also bumps the count
__global__ void scatter_accum4(const float* __restrict__ x,
                               const int* __restrict__ ei, // [2,E]
                               float* __restrict__ agg,    // [N,F]
                               float* __restrict__ cnt,    // [N]
                               int E, int Fq /* F/4 */) {
    long long tid = (long long)blockIdx.x * blockDim.x + threadIdx.x;
    long long total = (long long)E * Fq;
    if (tid >= total) return;
    int e = (int)(tid / Fq);
    int c = (int)(tid % Fq);
    int src = ei[e];
    int dst = ei[E + e];
    float4 v = ((const float4*)x)[(long long)src * Fq + c];
    float* dp = agg + ((long long)dst * Fq + c) * 4;
    atomicAdd(dp + 0, v.x);
    atomicAdd(dp + 1, v.y);
    atomicAdd(dp + 2, v.z);
    atomicAdd(dp + 3, v.w);
    if (c == 0) atomicAdd(&cnt[dst], 1.0f);
}

// ---------------------------------------------------------------- bf16 pack
__device__ __forceinline__ void pack16(v16bf& d, float4 a, float4 b,
                                       float4 c, float4 e) {
    d[0]  = (__bf16)a.x; d[1]  = (__bf16)a.y; d[2]  = (__bf16)a.z; d[3]  = (__bf16)a.w;
    d[4]  = (__bf16)b.x; d[5]  = (__bf16)b.y; d[6]  = (__bf16)b.z; d[7]  = (__bf16)b.w;
    d[8]  = (__bf16)c.x; d[9]  = (__bf16)c.y; d[10] = (__bf16)c.z; d[11] = (__bf16)c.w;
    d[12] = (__bf16)e.x; d[13] = (__bf16)e.y; d[14] = (__bf16)e.z; d[15] = (__bf16)e.w;
}

// ------------------------------------------------------ fused SAGE-conv GEMM
// out[m,n] = relu( bias[n] + (1/max(cnt[m],1)) * (agg@wl^T)[m,n] + (x@wr^T)[m,n] )
// One wave per 16x16 output tile; two independent bf16 WMMA accumulator chains
// (no intra-iteration WMMA->WMMA RAW); all operand loads are b128.
// Requires: K multiple of 32, N multiple of 16. Ragged M handled via row clamp.
__global__ void sage_gemm_wmma(const float* __restrict__ agg,
                               const float* __restrict__ cnt,
                               const float* __restrict__ xin,
                               const float* __restrict__ wl,   // [N,K]
                               const float* __restrict__ wr,   // [N,K]
                               const float* __restrict__ bias, // [N]
                               float* __restrict__ out,        // [M,N]
                               int M, int K, int N) {
    const int lane   = threadIdx.x & 31;
    const int ntiles = N >> 4;
    const int mtile  = blockIdx.x / ntiles;
    const int ntile  = blockIdx.x % ntiles;
    const int hi     = lane >> 4;        // half-wave select

    int row = mtile * 16 + (lane & 15);
    if (row >= M) row = M - 1;           // clamp: keeps loads valid, no EXEC games
    const int col = ntile * 16 + (lane & 15);

    const float* aRow = agg + (long long)row * K;
    const float* xRow = xin + (long long)row * K;
    const float* lRow = wl  + (long long)col * K;
    const float* rRow = wr  + (long long)col * K;

    v8f accA = {0.f,0.f,0.f,0.f,0.f,0.f,0.f,0.f};
    v8f accX = {0.f,0.f,0.f,0.f,0.f,0.f,0.f,0.f};

    for (int k0 = 0; k0 < K; k0 += 32) {
        const int kA = k0 + hi * 16;     // A lane chunk: [kA..kA+15]
        const int kB = k0 + hi * 16;     // B lane chunk: [kB..kB+15]

        float4 a0 = *(const float4*)(aRow + kA);
        float4 a1 = *(const float4*)(aRow + kA + 4);
        float4 a2 = *(const float4*)(aRow + kA + 8);
        float4 a3 = *(const float4*)(aRow + kA + 12);
        float4 x0 = *(const float4*)(xRow + kA);
        float4 x1 = *(const float4*)(xRow + kA + 4);
        float4 x2 = *(const float4*)(xRow + kA + 8);
        float4 x3 = *(const float4*)(xRow + kA + 12);
        float4 l0 = *(const float4*)(lRow + kB);
        float4 l1 = *(const float4*)(lRow + kB + 4);
        float4 l2 = *(const float4*)(lRow + kB + 8);
        float4 l3 = *(const float4*)(lRow + kB + 12);
        float4 r0 = *(const float4*)(rRow + kB);
        float4 r1 = *(const float4*)(rRow + kB + 4);
        float4 r2 = *(const float4*)(rRow + kB + 8);
        float4 r3 = *(const float4*)(rRow + kB + 12);

        v16bf aA, aX, bL, bR;
        pack16(aA, a0, a1, a2, a3);
        pack16(aX, x0, x1, x2, x3);
        pack16(bL, l0, l1, l2, l3);
        pack16(bR, r0, r1, r2, r3);

        accA = __builtin_amdgcn_wmma_f32_16x16x32_bf16(false, aA, false, bL,
                                                       (short)0, accA, false, false);
        accX = __builtin_amdgcn_wmma_f32_16x16x32_bf16(false, aX, false, bR,
                                                       (short)0, accX, false, false);
    }

    // C/D layout: VGPR v -> M = v + 8*hi, N = lane&15
    const float bv = bias[col];
#pragma unroll
    for (int v = 0; v < 8; ++v) {
        int m = mtile * 16 + v + (hi << 3);
        if (m < M) {
            float invm = 1.0f / fmaxf(cnt[m], 1.0f);
            out[(long long)m * N + col] = fmaxf(accA[v] * invm + accX[v] + bv, 0.0f);
        }
    }
}

// --------------------------------------------- last-column distance (sort key)
__global__ void lastcol_kernel(const float* __restrict__ out1,
                               const float* __restrict__ out2,
                               float* __restrict__ lastcol,
                               int M, int lastRow) {
    int m = blockIdx.x * blockDim.x + threadIdx.x;
    if (m >= M) return;
    const float4* a = (const float4*)(out1 + (long long)m * H2DIM);
    const float4* b = (const float4*)(out2 + (long long)lastRow * H2DIM);
    float s = 0.0f;
#pragma unroll
    for (int k = 0; k < H2DIM / 4; ++k) {
        float4 av = a[k], bv = b[k];
        float dx = av.x - bv.x, dy = av.y - bv.y, dz = av.z - bv.z, dw = av.w - bv.w;
        s += dx * dx + dy * dy + dz * dz + dw * dw;
    }
    lastcol[m] = sqrtf(fmaxf(s, 0.0f));
}

// ------------------------------- per-graph top-50 selection (desc, stable ties)
__global__ void topk_select(const float* __restrict__ lastcol,
                            int* __restrict__ sel /* [NGRAPH,TOPK] */) {
    __shared__ float sv[NPG];
    __shared__ float rv[256];
    __shared__ int   ri[256];
    int b = blockIdx.x, t = threadIdx.x;
    for (int i = t; i < NPG; i += blockDim.x) sv[i] = lastcol[b * NPG + i];
    __syncthreads();
    for (int k = 0; k < TOPK; ++k) {
        float best = -INFINITY; int bi = 0x7fffffff;
        for (int i = t; i < NPG; i += blockDim.x) {
            float v = sv[i];
            if (v > best || (v == best && i < bi)) { best = v; bi = i; }
        }
        rv[t] = best; ri[t] = bi;
        __syncthreads();
        for (int s = 128; s > 0; s >>= 1) {
            if (t < s) {
                if (rv[t + s] > rv[t] || (rv[t + s] == rv[t] && ri[t + s] < ri[t])) {
                    rv[t] = rv[t + s]; ri[t] = ri[t + s];
                }
            }
            __syncthreads();
        }
        if (t == 0) { sel[b * TOPK + k] = b * NPG + ri[0]; sv[ri[0]] = -INFINITY; }
        __syncthreads();
    }
}

// --------------------------------------- distances for the selected 500 rows
__global__ void topk_rows(const float* __restrict__ out1,
                          const float* __restrict__ out2,
                          const int* __restrict__ sel,
                          float* __restrict__ topk, // [NGRAPH, TOPK*N2]
                          int N2) {
    int idx = blockIdx.x * blockDim.x + threadIdx.x;
    int total = NGRAPH * TOPK * N2;
    if (idx >= total) return;
    int n  = idx % N2;
    int bk = idx / N2;
    int node = sel[bk];
    const float4* a = (const float4*)(out1 + (long long)node * H2DIM);
    const float4* b = (const float4*)(out2 + (long long)n * H2DIM);
    float s = 0.0f;
#pragma unroll
    for (int k = 0; k < H2DIM / 4; ++k) {
        float4 av = a[k], bv = b[k];
        float dx = av.x - bv.x, dy = av.y - bv.y, dz = av.z - bv.z, dw = av.w - bv.w;
        s += dx * dx + dy * dy + dz * dz + dw * dw;
    }
    topk[idx] = sqrtf(fmaxf(s, 0.0f));
}

// ------------------------------------------------- fused MLP head (10 blocks)
__global__ void head_kernel(const float* __restrict__ topk, // [NGRAPH, KN2]
                            const float* __restrict__ fc1_w, const float* __restrict__ fc1_b,
                            const float* __restrict__ ln1_g, const float* __restrict__ ln1_b,
                            const float* __restrict__ fc2_w, const float* __restrict__ fc2_b,
                            const float* __restrict__ ln2_g, const float* __restrict__ ln2_b,
                            const float* __restrict__ fc3_w, const float* __restrict__ fc3_b,
                            float* __restrict__ out, int KN2) {
    __shared__ float h1s[128];
    __shared__ float h2s[64];
    __shared__ float red[128];
    int b = blockIdx.x, t = threadIdx.x;

    // fc1: 128 outputs, one per thread (KN2 = 9950 is even -> float2)
    const float2* xrow = (const float2*)(topk + (long long)b * KN2);
    const float2* wrow = (const float2*)(fc1_w + (long long)t * KN2);
    float acc = fc1_b[t];
    for (int i = 0; i < KN2 / 2; ++i) {
        float2 xv = xrow[i], wv = wrow[i];
        acc += xv.x * wv.x + xv.y * wv.y;
    }

    // LayerNorm(128) + ReLU
    red[t] = acc; __syncthreads();
    for (int s = 64; s > 0; s >>= 1) { if (t < s) red[t] += red[t + s]; __syncthreads(); }
    float mu = red[0] * (1.0f / 128.0f); __syncthreads();
    float d = acc - mu; red[t] = d * d; __syncthreads();
    for (int s = 64; s > 0; s >>= 1) { if (t < s) red[t] += red[t + s]; __syncthreads(); }
    float var = red[0] * (1.0f / 128.0f); __syncthreads();
    h1s[t] = fmaxf((acc - mu) * rsqrtf(var + 1e-5f) * ln1_g[t] + ln1_b[t], 0.0f);
    __syncthreads();

    // fc2: 64 outputs on threads 0..63
    float acc2 = 0.0f;
    if (t < 64) {
        acc2 = fc2_b[t];
        const float* w2 = fc2_w + t * 128;
        for (int i = 0; i < 128; ++i) acc2 += h1s[i] * w2[i];
    }
    __syncthreads();

    // LayerNorm(64) + ReLU
    red[t] = (t < 64) ? acc2 : 0.0f; __syncthreads();
    for (int s = 64; s > 0; s >>= 1) { if (t < s) red[t] += red[t + s]; __syncthreads(); }
    float mu2 = red[0] * (1.0f / 64.0f); __syncthreads();
    float d2 = (t < 64) ? (acc2 - mu2) : 0.0f; red[t] = d2 * d2; __syncthreads();
    for (int s = 64; s > 0; s >>= 1) { if (t < s) red[t] += red[t + s]; __syncthreads(); }
    float var2 = red[0] * (1.0f / 64.0f); __syncthreads();
    if (t < 64)
        h2s[t] = fmaxf((acc2 - mu2) * rsqrtf(var2 + 1e-5f) * ln2_g[t] + ln2_b[t], 0.0f);
    __syncthreads();

    // fc3 + sigmoid
    if (t == 0) {
        float a3 = fc3_b[0];
        for (int i = 0; i < 64; ++i) a3 += h2s[i] * fc3_w[i];
        out[b] = 1.0f / (1.0f + expf(-a3));
    }
}

// ---------------------------------------------------------------------------
extern "C" void kernel_launch(void* const* d_in, const int* in_sizes, int n_in,
                              void* d_out, int out_size, void* d_ws, size_t ws_size,
                              hipStream_t stream) {
    const float* x1  = (const float*)d_in[0];
    const int*   ei1 = (const int*)d_in[1];
    // d_in[2] = batch1 (equal-size graphs -> unused)
    const float* x2  = (const float*)d_in[3];
    const int*   ei2 = (const int*)d_in[4];
    const float* w1l = (const float*)d_in[5];
    const float* b1l = (const float*)d_in[6];
    const float* w1r = (const float*)d_in[7];
    const float* w2l = (const float*)d_in[8];
    const float* b2l = (const float*)d_in[9];
    const float* w2r = (const float*)d_in[10];
    const float* fc1_w = (const float*)d_in[11];
    const float* fc1_b = (const float*)d_in[12];
    const float* ln1_g = (const float*)d_in[13];
    const float* ln1_b = (const float*)d_in[14];
    const float* fc2_w = (const float*)d_in[15];
    const float* fc2_b = (const float*)d_in[16];
    const float* ln2_g = (const float*)d_in[17];
    const float* ln2_b = (const float*)d_in[18];
    const float* fc3_w = (const float*)d_in[19];
    const float* fc3_b = (const float*)d_in[20];
    float* out = (float*)d_out;

    const int M1 = in_sizes[0] / F_IN;       // 50000
    const int E1 = in_sizes[1] / 2;          // 1e6
    const int N2 = in_sizes[3] / F_IN;       // 199
    const int E2 = in_sizes[4] / 2;          // 4000
    const int KN2 = TOPK * N2;               // 9950

    // workspace carve-up
    char* wsp = (char*)d_ws;
    size_t off = 0;
    auto carve = [&](size_t bytes) -> void* {
        void* p = wsp + off;
        off += (bytes + 255) & ~(size_t)255;
        return p;
    };
    float* agg   = (float*)carve((size_t)M1 * F_IN * 4);
    float* cnt   = (float*)carve((size_t)M1 * 4);
    float* h1    = (float*)carve((size_t)M1 * H1DIM * 4);
    float* o1    = (float*)carve((size_t)M1 * H2DIM * 4);
    float* agg2  = (float*)carve((size_t)N2 * F_IN * 4);
    float* cnt2  = (float*)carve((size_t)N2 * 4);
    float* h2    = (float*)carve((size_t)N2 * H1DIM * 4);
    float* o2    = (float*)carve((size_t)N2 * H2DIM * 4);
    float* lastc = (float*)carve((size_t)M1 * 4);
    int*   sel   = (int*)  carve((size_t)NGRAPH * TOPK * 4);
    float* tkd   = (float*)carve((size_t)NGRAPH * KN2 * 4);
    (void)ws_size; (void)n_in; (void)out_size;

    const int ZB = 2048;
    auto launch_zero = [&](float* p, long long n) {
        zero_kernel<<<ZB, 256, 0, stream>>>(p, n);
    };
    auto launch_scatter = [&](const float* x, const int* ei, float* a, float* c,
                              int E, int F) {
        long long total = (long long)E * (F / 4);
        unsigned blocks = (unsigned)((total + 255) / 256);
        scatter_accum4<<<blocks, 256, 0, stream>>>(x, ei, a, c, E, F / 4);
    };
    auto launch_gemm = [&](const float* a, const float* c, const float* x,
                           const float* wl, const float* wr, const float* bias,
                           float* o, int M, int K, int N) {
        int mtiles = (M + 15) >> 4, ntiles = N >> 4;
        sage_gemm_wmma<<<mtiles * ntiles, 32, 0, stream>>>(a, c, x, wl, wr, bias,
                                                           o, M, K, N);
    };

    // ---------------- graph 1 GNN ----------------
    launch_zero(agg, (long long)M1 * F_IN);
    launch_zero(cnt, M1);
    launch_scatter(x1, ei1, agg, cnt, E1, F_IN);
    launch_gemm(agg, cnt, x1, w1l, w1r, b1l, h1, M1, F_IN, H1DIM);

    launch_zero(agg, (long long)M1 * H1DIM);
    launch_zero(cnt, M1);
    launch_scatter(h1, ei1, agg, cnt, E1, H1DIM);
    launch_gemm(agg, cnt, h1, w2l, w2r, b2l, o1, M1, H1DIM, H2DIM);

    // ---------------- graph 2 GNN ----------------
    launch_zero(agg2, (long long)N2 * F_IN);
    launch_zero(cnt2, N2);
    launch_scatter(x2, ei2, agg2, cnt2, E2, F_IN);
    launch_gemm(agg2, cnt2, x2, w1l, w1r, b1l, h2, N2, F_IN, H1DIM);

    launch_zero(agg2, (long long)N2 * H1DIM);
    launch_zero(cnt2, N2);
    launch_scatter(h2, ei2, agg2, cnt2, E2, H1DIM);
    launch_gemm(agg2, cnt2, h2, w2l, w2r, b2l, o2, N2, H1DIM, H2DIM);

    // ---------------- sort key + top-k selection ----------------
    lastcol_kernel<<<(M1 + 255) / 256, 256, 0, stream>>>(o1, o2, lastc, M1, N2 - 1);
    topk_select<<<NGRAPH, 256, 0, stream>>>(lastc, sel);

    int totalTk = NGRAPH * TOPK * N2;
    topk_rows<<<(totalTk + 255) / 256, 256, 0, stream>>>(o1, o2, sel, tkd, N2);

    // ---------------- MLP head ----------------
    head_kernel<<<NGRAPH, 128, 0, stream>>>(tkd, fc1_w, fc1_b, ln1_g, ln1_b,
                                            fc2_w, fc2_b, ln2_g, ln2_b,
                                            fc3_w, fc3_b, out, KN2);
}
